// RecursiveTRM_9809705304256
// MI455X (gfx1250) — compile-verified, matrix operand order checked by reading, not run on hardware
//
#include <hip/hip_runtime.h>
#include <hip/hip_bf16.h>

// ---------------------------------------------------------------------------
// RecursiveTRM persistent RNN kernel for MI455X (gfx1250, wave32, WMMA).
//
//  * Batch rows are independent -> 4 persistent workgroups, each owning a
//    16-row batch slice (M=16 == one WMMA tile). All ~7200 dependent GEMM
//    stages (512 t x 2 layers x (GRU + 3 refinements)) sync via workgroup
//    barriers in one WGP -- never a grid-wide sync.
//  * All GEMMs: v_wmma_f32_16x16x32_bf16 (f32 accumulate). Weights pre-packed
//    once into per-lane WMMA B-fragment order (two global b128 per fragment);
//    ~2.9 MB bf16 total -> permanently L2-resident.
//  * Activations live in LDS bf16 row-major (A fragments = two ds_load_b128).
//    Elementwise recurrent state (h, out) kept in registers per owning thread
//    (C-layout match) -> no LDS re-reads, better precision.
//  * Rolled refinement loop (#pragma unroll 1) + launch_bounds(512,1) keep
//    VGPR pressure inside budget: round-2 experiment showed that letting the
//    compiler unroll the K=3 loop spills staged B-fragments to scratch.
//  * x_{t+1} prefetched during compute; output stored non-temporal (NT) so
//    the ~34 MB stream does not evict L2-resident weights.
//  * 16 waves / WG (4 per SIMD32) to hide WMMA + LDS latency.
// ---------------------------------------------------------------------------

typedef __bf16 bf16;
typedef __attribute__((ext_vector_type(16))) __bf16 v16bf;
typedef __attribute__((ext_vector_type(8)))  float  v8f;

#define B_   64
#define T_   512
#define D_   256
#define H_   256
#define NL_  2
#define KR_  3
#define BM_  16        // batch rows per workgroup
#define NTHREADS_ 512  // 16 waves

// per-layer packed weight sizes (bf16 elements)
#define SZ_IH_ (48*8*512)   // (3H x H)  -> 48 n-tiles x 8 k-tiles x 512
#define SZ_G_  (16*16*512)  // (H x 2H)  -> 16 n-tiles x 16 k-tiles
#define SZ_U_  (16*8*512)   // (H x H)   -> 16 n-tiles x 8 k-tiles

__device__ __forceinline__ float fsigmoid(float x) {
  // 1 / (1 + 2^(-x*log2(e)))  -> v_exp_f32 + v_rcp_f32
  float e = __builtin_amdgcn_exp2f(-x * 1.44269504088896340736f);
  return __builtin_amdgcn_rcpf(1.0f + e);
}
__device__ __forceinline__ float ftanh(float x) {
  float e = __builtin_amdgcn_exp2f(-2.0f * x * 1.44269504088896340736f);
  return __builtin_amdgcn_rcpf(1.0f + e) * 2.0f - 1.0f;
}

union FragU { v16bf v; uint4 q[2]; };

// A fragment (16x32 bf16) from a row-major bf16 activation buffer in LDS.
// Lane layout (ISA 7.12.2): M = lane&15; half-wave hi=lane>>4 selects K runs
// {hi*8..hi*8+7} and {16+hi*8..16+hi*8+7} -> two contiguous 16B chunks.
__device__ __forceinline__ v16bf load_a_frag(const bf16* base, int stride,
                                             int ktile, int lane) {
  const int m  = lane & 15;
  const int hi = lane >> 4;
  const bf16* p0 = base + m * stride + ktile * 32 + hi * 8;
  FragU u;
  u.q[0] = *(const uint4*)p0;
  u.q[1] = *(const uint4*)(p0 + 16);
  return u.v;
}

// B fragment (32x16 bf16) from the pre-packed weight stream: 32 contiguous
// bytes per lane (two global b128 loads), fragment = 1KB.
__device__ __forceinline__ v16bf load_b_frag(const bf16* pk, int frag, int lane) {
  const uint4* p = (const uint4*)(pk + (size_t)frag * 512) + lane * 2;
  FragU u;
  u.q[0] = p[0];
  u.q[1] = p[1];
  return u.v;
}

__device__ __forceinline__ v8f wmma_bf16(v16bf a, v16bf b, v8f c) {
  return __builtin_amdgcn_wmma_f32_16x16x32_bf16(
      /*neg_a=*/false, a, /*neg_b=*/false, b,
      /*c_mod=*/(short)0, c, /*reuse_a=*/false, /*reuse_b=*/false);
}

// ---------------------------------------------------------------------------
// Pack fp32 weight matrix W (N x Kd, row-major; GEMM does act @ W^T, so the
// WMMA B matrix is W^T: B[k][n] = W[n][k]) into per-lane fragment order.
// dst element index = ((ntile*KT + ktile)*32 + lane)*16 + e
// ---------------------------------------------------------------------------
__global__ void pack_wmma_b(const float* __restrict__ src, bf16* __restrict__ dst,
                            int N, int Kd) {
  const int KT = Kd / 32;
  const int total = (N / 16) * KT * 512;
  for (int i = blockIdx.x * blockDim.x + threadIdx.x; i < total;
       i += gridDim.x * blockDim.x) {
    const int e     = i & 15;
    const int lane  = (i >> 4) & 31;
    const int frag  = i >> 9;
    const int ktile = frag % KT;
    const int ntile = frag / KT;
    const int n  = ntile * 16 + (lane & 15);
    const int hi = lane >> 4;
    const int kk = e + hi * 8 + (e >= 8 ? 8 : 0);
    const int k  = ktile * 32 + kk;
    dst[i] = (bf16)src[(size_t)n * Kd + k];
  }
}

// ---------------------------------------------------------------------------
// Persistent RNN kernel: grid = 4 blocks x 512 threads (16 waves).
// Wave w owns output column tile w (H space); for the GRU it also owns the
// matching r/z/n tiles (w, w+16, w+32 in 3H space) so gate math is local.
// C layout: acc[r] of lane L holds element (M = r + 8*(L>>4), N = L&15).
// ---------------------------------------------------------------------------
__global__ __launch_bounds__(NTHREADS_, 1) void rnn_persistent(
    const float* __restrict__ x,
    const float* __restrict__ b_ih, const float* __restrict__ b_hh,
    const float* __restrict__ bg,   const float* __restrict__ bt,
    const float* __restrict__ bu,
    const bf16* __restrict__ pkWih, const bf16* __restrict__ pkWhh,
    const bf16* __restrict__ pkWg,  const bf16* __restrict__ pkWt,
    const bf16* __restrict__ pkWu,
    float* __restrict__ out) {
  __shared__ alignas(16) bf16 sH[NL_][BM_][H_];     // layer hidden states
  __shared__ alignas(16) bf16 sIn[BM_][D_];         // layer-0 input (x_t)
  __shared__ alignas(16) bf16 sComb[BM_][2 * H_];   // [out | rh]

  const int tid     = threadIdx.x;
  const int lane    = tid & 31;
  const int w       = tid >> 5;        // wave id 0..15
  const int rowbase = blockIdx.x * BM_;
  const int ncol    = lane & 15;
  const int hi      = lane >> 4;
  const int col     = w * 16 + ncol;   // owned H-column of this lane

  // register-resident recurrent state for the owned C-tile elements
  float hreg[NL_][8];
  #pragma unroll
  for (int l = 0; l < NL_; ++l)
    #pragma unroll
    for (int r = 0; r < 8; ++r) hreg[l][r] = 0.0f;

  for (int i = tid; i < NL_ * BM_ * H_; i += NTHREADS_)
    (&sH[0][0][0])[i] = (bf16)0.0f;
  __syncthreads();

  #pragma unroll 1
  for (int t = 0; t < T_; ++t) {
    // stage x_t slice (16 x 256) as bf16 into LDS; prefetch x_{t+1}
    for (int i = tid; i < BM_ * D_; i += NTHREADS_) {
      const int m = i >> 8, d = i & 255;
      const size_t base = ((size_t)(rowbase + m) * T_ + t) * D_ + d;
      (&sIn[0][0])[i] = (bf16)x[base];
      if (t + 1 < T_) __builtin_prefetch(&x[base + D_], 0, 0);
    }
    __syncthreads();

    float onew[8];

    #pragma unroll 1
    for (int l = 0; l < NL_; ++l) {
      const bf16* Ain     = (l == 0) ? &sIn[0][0] : &sComb[0][0];
      const int   Astride = (l == 0) ? D_ : 2 * H_;
      const bf16* Wih = pkWih + (size_t)l * SZ_IH_;
      const bf16* Whh = pkWhh + (size_t)l * SZ_IH_;
      const bf16* Wg_ = pkWg  + (size_t)l * SZ_G_;
      const bf16* Wt_ = pkWt  + (size_t)l * SZ_G_;
      const bf16* Wu_ = pkWu  + (size_t)l * SZ_U_;

      // ---- GRU: gi = inp@Wih^T, gh = h@Whh^T, tiles (w, w+16, w+32) ----
      v8f gi[3], gh[3];
      #pragma unroll
      for (int g3 = 0; g3 < 3; ++g3) {
        const int ntile = w + 16 * g3;
        v8f acc = {};
        #pragma unroll
        for (int kt = 0; kt < 8; ++kt)
          acc = wmma_bf16(load_a_frag(Ain, Astride, kt, lane),
                          load_b_frag(Wih, ntile * 8 + kt, lane), acc);
        gi[g3] = acc;
        acc = (v8f){};
        #pragma unroll
        for (int kt = 0; kt < 8; ++kt)
          acc = wmma_bf16(load_a_frag(&sH[l][0][0], H_, kt, lane),
                          load_b_frag(Whh, ntile * 8 + kt, lane), acc);
        gh[g3] = acc;
      }

      const float bir = b_ih[l * 3 * H_ + col];
      const float biz = b_ih[l * 3 * H_ + H_ + col];
      const float bin = b_ih[l * 3 * H_ + 2 * H_ + col];
      const float bhr = b_hh[l * 3 * H_ + col];
      const float bhz = b_hh[l * 3 * H_ + H_ + col];
      const float bhn = b_hh[l * 3 * H_ + 2 * H_ + col];

      float ocur[8];  // running 'out' in registers (exact, no bf16 trip)
      #pragma unroll
      for (int r = 0; r < 8; ++r) {
        const float rr = fsigmoid(gi[0][r] + bir + gh[0][r] + bhr);
        const float zz = fsigmoid(gi[1][r] + biz + gh[1][r] + bhz);
        const float nn = ftanh(gi[2][r] + bin + rr * (gh[2][r] + bhn));
        const float hn = (1.0f - zz) * nn + zz * hreg[l][r];
        hreg[l][r] = hn;
        ocur[r]    = hn;
      }
      __syncthreads();  // all WMMA reads of sH / layer input complete
      #pragma unroll
      for (int r = 0; r < 8; ++r) {
        const int m = r + hi * 8;
        const bf16 hb = (bf16)hreg[l][r];
        sH[l][m][col]        = hb;  // new hidden (A-matrix for next t's gh)
        sComb[m][col]        = hb;  // out = h_new
        sComb[m][H_ + col]   = hb;  // rh  = h_new
      }
      __syncthreads();

      // ---- K=3 refinement iterations (kept rolled: bounds VGPR pressure) --
      const float bgv = bg[l * H_ + col];
      const float btv = bt[l * H_ + col];
      const float buv = bu[l * H_ + col];
      #pragma unroll 1
      for (int it = 0; it < KR_; ++it) {
        v8f ag = {}, ar = {};
        #pragma unroll
        for (int kt = 0; kt < 16; ++kt) {   // K = 2H = 512 over [out|rh]
          const v16bf a = load_a_frag(&sComb[0][0], 2 * H_, kt, lane);
          ag = wmma_bf16(a, load_b_frag(Wg_, w * 16 + kt, lane), ag);
          ar = wmma_bf16(a, load_b_frag(Wt_, w * 16 + kt, lane), ar);
        }
        #pragma unroll
        for (int r = 0; r < 8; ++r) {
          const float gv = fsigmoid(ag[r] + bgv);
          const float rv = ftanh(ar[r] + btv);
          onew[r] = gv * rv + (1.0f - gv) * ocur[r];
          ocur[r] = onew[r];
        }
        __syncthreads();  // comb reads complete before overwrite
        #pragma unroll
        for (int r = 0; r < 8; ++r)
          sComb[(r + hi * 8)][col] = (bf16)onew[r];
        __syncthreads();
        // rh = tanh(out_new @ Wu^T + bu); A = first half of comb (stride 2H)
        v8f au = {};
        #pragma unroll
        for (int kt = 0; kt < 8; ++kt)
          au = wmma_bf16(load_a_frag(&sComb[0][0], 2 * H_, kt, lane),
                         load_b_frag(Wu_, w * 8 + kt, lane), au);
        #pragma unroll
        for (int r = 0; r < 8; ++r)
          sComb[(r + hi * 8)][H_ + col] = (bf16)ftanh(au[r] + buv);
        __syncthreads();
      }
      // stage output (next layer's input) = sComb first half
    }  // layers

    // final layer's out -> output[B][T][H] (fp32), streamed non-temporal so
    // it does not evict the L2-resident weight working set
    #pragma unroll
    for (int r = 0; r < 8; ++r) {
      const int m = r + hi * 8;
      __builtin_nontemporal_store(
          onew[r], &out[((size_t)(rowbase + m) * T_ + t) * H_ + col]);
    }
  }    // timesteps

  // final hidden: [L][B][H] appended after output (exact register values)
  #pragma unroll
  for (int l = 0; l < NL_; ++l)
    #pragma unroll
    for (int r = 0; r < 8; ++r) {
      const int m = r + hi * 8;
      out[(size_t)B_ * T_ * H_ + (size_t)l * B_ * H_ +
          (size_t)(rowbase + m) * H_ + col] = hreg[l][r];
    }
}

// ---------------------------------------------------------------------------
extern "C" void kernel_launch(void* const* d_in, const int* in_sizes, int n_in,
                              void* d_out, int out_size, void* d_ws, size_t ws_size,
                              hipStream_t stream) {
  (void)in_sizes; (void)n_in; (void)out_size; (void)ws_size;
  const float* x    = (const float*)d_in[0];
  const float* W_ih = (const float*)d_in[1];
  const float* b_ih = (const float*)d_in[2];
  const float* W_hh = (const float*)d_in[3];
  const float* b_hh = (const float*)d_in[4];
  const float* Wg   = (const float*)d_in[5];
  const float* bg   = (const float*)d_in[6];
  const float* Wt   = (const float*)d_in[7];
  const float* bt   = (const float*)d_in[8];
  const float* Wu   = (const float*)d_in[9];
  const float* bu   = (const float*)d_in[10];
  float* out = (float*)d_out;

  // workspace: packed bf16 weights (~2.9 MB total)
  bf16* pkWih = (bf16*)d_ws;
  bf16* pkWhh = pkWih + (size_t)NL_ * SZ_IH_;
  bf16* pkWg  = pkWhh + (size_t)NL_ * SZ_IH_;
  bf16* pkWt  = pkWg  + (size_t)NL_ * SZ_G_;
  bf16* pkWu  = pkWt  + (size_t)NL_ * SZ_G_;

  for (int l = 0; l < NL_; ++l) {
    pack_wmma_b<<<(SZ_IH_ + 255) / 256, 256, 0, stream>>>(
        W_ih + (size_t)l * 768 * 256, pkWih + (size_t)l * SZ_IH_, 768, 256);
    pack_wmma_b<<<(SZ_IH_ + 255) / 256, 256, 0, stream>>>(
        W_hh + (size_t)l * 768 * 256, pkWhh + (size_t)l * SZ_IH_, 768, 256);
    pack_wmma_b<<<(SZ_G_ + 255) / 256, 256, 0, stream>>>(
        Wg + (size_t)l * 256 * 512, pkWg + (size_t)l * SZ_G_, 256, 512);
    pack_wmma_b<<<(SZ_G_ + 255) / 256, 256, 0, stream>>>(
        Wt + (size_t)l * 256 * 512, pkWt + (size_t)l * SZ_G_, 256, 512);
    pack_wmma_b<<<(SZ_U_ + 255) / 256, 256, 0, stream>>>(
        Wu + (size_t)l * 256 * 256, pkWu + (size_t)l * SZ_U_, 256, 256);
  }

  // 4 persistent workgroups (one 16-row batch slice each), 16 waves/WG
  rnn_persistent<<<dim3(B_ / BM_), dim3(NTHREADS_), 0, stream>>>(
      x, b_ih, b_hh, bg, bt, bu, pkWih, pkWhh, pkWg, pkWt, pkWu, out);
}
